// MultiHeadSelfAttention_10531259810702
// MI455X (gfx1250) — matrix operand order repaired and checked by reading, not compiled, not run on hardware
//
#include <hip/hip_runtime.h>
#include <hip/hip_bf16.h>

// Problem constants (match reference)
#define B_    4
#define T_    2048
#define DM    1024
#define NH    16
#define DKH   64
#define BT    (B_ * T_)      // 8192 rows
#define NQKV  (3 * DM)       // 3072

typedef __bf16 bf16_t;
typedef __attribute__((ext_vector_type(16))) __bf16 v16bf;
typedef __attribute__((ext_vector_type(8)))  __bf16 v8bf;
typedef __attribute__((ext_vector_type(4)))  __bf16 v4bf;
typedef __attribute__((ext_vector_type(8)))  float  v8f;

union BF16Frag { v16bf v; v8bf h[2]; };

__device__ __forceinline__ v8f zero8() {
    v8f z = {0.f,0.f,0.f,0.f,0.f,0.f,0.f,0.f};
    return z;
}

// WGP-scope prefetch (scope 0 is the default encoding: no scope modifier).
// Pulls into all cache levels on miss, stops on WGP hit — unlike
// __builtin_prefetch which lowers to SE/SYS scope (GL2 only).
__device__ __forceinline__ void prefetch_wgp(const void* p) {
    asm volatile("global_prefetch_b8 %0, off" :: "v"(p) : "memory");
}

// A-matrix 16x32 bf16 fragment (ISA 7.12.2): lane L holds row M=L%16;
// VGPR0-3 <- K = 8*(L/16)+0..7 ; VGPR4-7 <- K = 16+8*(L/16)+0..7
__device__ __forceinline__ v16bf load_a_frag(const bf16_t* base, int stride, int kbase, int lane) {
    const int half = lane >> 4;
    const int m    = lane & 15;
    const bf16_t* p = base + (size_t)m * stride + kbase;
    BF16Frag f;
    f.h[0] = *(const v8bf*)(p + half * 8);
    f.h[1] = *(const v8bf*)(p + 16 + half * 8);
    return f.v;
}

// B-matrix 32x16 bf16 fragment: lane L holds column N=L%16;
// VGPR0-3 <- K = 8*(L/16)+0..7 ; VGPR4-7 <- K = 16+8*(L/16)+0..7. `base` points at B-transposed storage
// (row n of base == column n of B), so this is two 16B loads.
__device__ __forceinline__ v16bf load_b_frag(const bf16_t* base, int stride, int kbase, int lane) {
    const int half = lane >> 4;
    const int n    = lane & 15;
    const bf16_t* p = base + (size_t)n * stride + kbase;
    BF16Frag f;
    f.h[0] = *(const v8bf*)(p + half * 8);
    f.h[1] = *(const v8bf*)(p + 16 + half * 8);
    return f.v;
}

__device__ __forceinline__ v8f wmma_bf16f32(v16bf a, v16bf b, v8f c) {
    return __builtin_amdgcn_wmma_f32_16x16x32_bf16(false, a, false, b, (short)0, c, false, false);
}

// ---------------------------------------------------------------- conversions
__global__ void __launch_bounds__(256) cvt_f32_bf16_kernel(const float4* __restrict__ in,
                                                           v4bf* __restrict__ out, int n4) {
    int i = blockIdx.x * 256 + threadIdx.x;
    if (i < n4) {
        float4 f = in[i];
        v4bf r = { (bf16_t)f.x, (bf16_t)f.y, (bf16_t)f.z, (bf16_t)f.w };
        out[i] = r;
    }
}

// in: [RS, CS] f32 -> out: [CS, RS] bf16, 32x32 LDS tiles, coalesced both sides
template <int RS, int CS>
__global__ void __launch_bounds__(256) transpose_cvt_tiled(const float* __restrict__ in,
                                                           bf16_t* __restrict__ out) {
    __shared__ float tile[32][33];
    const int tx = threadIdx.x & 31;
    const int ty = threadIdx.x >> 5;              // 0..7
    const int c0 = blockIdx.x * 32;
    const int r0 = blockIdx.y * 32;
    #pragma unroll
    for (int j = ty; j < 32; j += 8)
        tile[j][tx] = in[(size_t)(r0 + j) * CS + (c0 + tx)];
    __syncthreads();
    #pragma unroll
    for (int j = ty; j < 32; j += 8)
        out[(size_t)(c0 + j) * RS + (r0 + tx)] = (bf16_t)tile[tx][j];
}

// ------------------------------------------------------- 32x64 GEMM core
// Ping-pong software pipeline (no register rotation copies) + full unroll
// (K/64 = 16 double-steps -> all addresses become immediate offsets) +
// WGP-scope global_prefetch of the panels two double-steps ahead.
__device__ __forceinline__ void load_abset(const bf16_t* Ar0, const bf16_t* Ar1,
                                           const bf16_t* Bc, int k, int lane,
                                           v16bf& a0, v16bf& a1, v16bf b[4]) {
    a0 = load_a_frag(Ar0, DM, k, lane);
    a1 = load_a_frag(Ar1, DM, k, lane);
    #pragma unroll
    for (int nt = 0; nt < 4; ++nt)
        b[nt] = load_b_frag(Bc + (size_t)nt * 16 * DM, DM, k, lane);
}

__device__ __forceinline__ void gemm_core_32x64(const bf16_t* __restrict__ A, int rowBase,
                                                const bf16_t* __restrict__ Bt, int colBase,
                                                v8f acc[2][4], int lane) {
    const bf16_t* Ar0 = A + (size_t)rowBase * DM;
    const bf16_t* Ar1 = Ar0 + (size_t)16 * DM;
    const bf16_t* Bc  = Bt + (size_t)colBase * DM;

    // per-lane prefetch bases: 32 A rows / 64 B rows covered by the wave
    const bf16_t* pfA = Ar0 + (size_t)lane * DM;
    const bf16_t* pfB = Bc  + (size_t)lane * 2 * DM;

    v16bf a0A, a1A, bA[4], a0B, a1B, bB[4];
    load_abset(Ar0, Ar1, Bc, 0, lane, a0A, a1A, bA);

    #pragma unroll
    for (int k = 0; k < DM; k += 64) {
        // fill buffer B for k+32 while computing buffer A
        load_abset(Ar0, Ar1, Bc, k + 32, lane, a0B, a1B, bB);
        int kpf = (k + 128 < DM) ? (k + 128) : k;
        prefetch_wgp(pfA + kpf);
        prefetch_wgp(pfB + kpf);
        #pragma unroll
        for (int nt = 0; nt < 4; ++nt) acc[0][nt] = wmma_bf16f32(a0A, bA[nt], acc[0][nt]);
        #pragma unroll
        for (int nt = 0; nt < 4; ++nt) acc[1][nt] = wmma_bf16f32(a1A, bA[nt], acc[1][nt]);

        // refill buffer A for k+64 while computing buffer B
        int k2 = (k + 64 < DM) ? (k + 64) : 0;     // last: harmless dummy
        load_abset(Ar0, Ar1, Bc, k2, lane, a0A, a1A, bA);
        #pragma unroll
        for (int nt = 0; nt < 4; ++nt) acc[0][nt] = wmma_bf16f32(a0B, bB[nt], acc[0][nt]);
        #pragma unroll
        for (int nt = 0; nt < 4; ++nt) acc[1][nt] = wmma_bf16f32(a1B, bB[nt], acc[1][nt]);
    }
}

// ---------------------------------------------------------------- QKV GEMM
// X[8192,1024]bf16 @ Wqkv^T-stored[3072,1024]bf16 -> scatter into Q,K,Vt
__global__ void __launch_bounds__(128) qkv_gemm_kernel(const bf16_t* __restrict__ X,
                                                       const bf16_t* __restrict__ Wt,
                                                       bf16_t* __restrict__ Q,
                                                       bf16_t* __restrict__ K,
                                                       bf16_t* __restrict__ Vt) {
    const int lane = threadIdx.x & 31;
    const int wave = threadIdx.x >> 5;
    const int NG = NQKV / 64;                       // 48 column groups
    int tile   = blockIdx.x * 4 + wave;             // 0 .. 256*48-1
    int mTile  = tile / NG;
    int nGroup = tile - mTile * NG;
    int rowBase = mTile * 32;
    int colBase = nGroup * 64;

    v8f acc[2][4];
    #pragma unroll
    for (int mt = 0; mt < 2; ++mt)
        #pragma unroll
        for (int nt = 0; nt < 4; ++nt) acc[mt][nt] = zero8();

    gemm_core_32x64(X, rowBase, Wt, colBase, acc, lane);

    // C/D layout: lane L, vgpr g -> M = g + 8*(L/16), N = L%16
    const int half = lane >> 4, nloc = lane & 15;
    #pragma unroll
    for (int nt = 0; nt < 4; ++nt) {
        int c = colBase + nt * 16 + nloc;
        int which = c >> 10;          // 0=q 1=k 2=v (uniform per nt)
        int cc = c & 1023;
        int h = cc >> 6;
        int d = cc & 63;
        #pragma unroll
        for (int mt = 0; mt < 2; ++mt) {
            #pragma unroll
            for (int g = 0; g < 8; ++g) {
                int r = rowBase + mt * 16 + g + half * 8;   // global token row
                int b = r >> 11, t = r & (T_ - 1);
                int bh = b * NH + h;
                float v = acc[mt][nt][g];
                if (which == 0)      Q[((size_t)bh * T_ + t) * DKH + d]  = (bf16_t)(v * 0.125f);
                else if (which == 1) K[((size_t)bh * T_ + t) * DKH + d]  = (bf16_t)v;
                else                 Vt[((size_t)bh * DKH + d) * T_ + t] = (bf16_t)v;
            }
        }
    }
}

// ---------------------------------------------------------------- attention
// 1 wave = 32 query rows of one (b,h); flash-style online softmax over
// 32-key blocks. S: 8 WMMAs, PV: 8 WMMAs per block. Next-block K frags and
// current V frags are issued before the softmax so they load under VALU work.
__global__ void __launch_bounds__(128) attn_kernel(const bf16_t* __restrict__ Q,
                                                   const bf16_t* __restrict__ K,
                                                   const bf16_t* __restrict__ Vt,
                                                   bf16_t* __restrict__ AttOut) {
    __shared__ __align__(16) bf16_t ldsP[4][32 * 32];

    const int lane = threadIdx.x & 31;
    const int wave = threadIdx.x >> 5;
    int wid   = blockIdx.x * 4 + wave;          // 0 .. 4095
    int mTile = wid & (T_ / 32 - 1);            // 64 m-tiles per (b,h)
    int bh    = wid >> 6;
    int b     = bh >> 4, h = bh & 15;

    const bf16_t* Qb = Q  + (size_t)bh * T_ * DKH;
    const bf16_t* Kb = K  + (size_t)bh * T_ * DKH;
    const bf16_t* Vb = Vt + (size_t)bh * DKH * T_;

    const int mBase = mTile * 32;
    const int half = lane >> 4, nloc = lane & 15;
    bf16_t* myP = ldsP[wave];

    // Q fragments: [m-tile][k-chunk], Q already scaled by 1/sqrt(dk)
    v16bf qf[2][2];
    #pragma unroll
    for (int mt = 0; mt < 2; ++mt)
        #pragma unroll
        for (int kc = 0; kc < 2; ++kc)
            qf[mt][kc] = load_a_frag(Qb + (size_t)(mBase + mt * 16) * DKH, DKH, kc * 32, lane);

    float m_i[2][8], l_i[2][8];
    v8f o[2][4];
    #pragma unroll
    for (int mt = 0; mt < 2; ++mt) {
        #pragma unroll
        for (int g = 0; g < 8; ++g) { m_i[mt][g] = -3.0e38f; l_i[mt][g] = 0.f; }
        #pragma unroll
        for (int c = 0; c < 4; ++c) o[mt][c] = zero8();
    }

    // preload K fragments for sBase = 0: [n-tile][k-chunk]
    v16bf kf[2][2];
    #pragma unroll
    for (int nt = 0; nt < 2; ++nt)
        #pragma unroll
        for (int kc = 0; kc < 2; ++kc)
            kf[nt][kc] = load_b_frag(Kb + (size_t)(nt * 16) * DKH, DKH, kc * 32, lane);

    const int sEnd = mBase + 32;                // causal: keys [0, sEnd)
    for (int sBase = 0; sBase < sEnd; sBase += 32) {
        // ---- S = Q K^T, 32x32 block as 2x2 16x16 tiles (8 WMMAs)
        v8f s[2][2];
        #pragma unroll
        for (int mt = 0; mt < 2; ++mt)
            #pragma unroll
            for (int nt = 0; nt < 2; ++nt) {
                v8f t = zero8();
                t = wmma_bf16f32(qf[mt][0], kf[nt][0], t);
                t = wmma_bf16f32(qf[mt][1], kf[nt][1], t);
                s[mt][nt] = t;
            }

        // ---- issue next-block K loads and current V loads early (overlap softmax)
        int sn = (sBase + 32 < sEnd) ? sBase + 32 : 0;   // dummy reload on last iter
        v16bf kfn[2][2];
        #pragma unroll
        for (int nt = 0; nt < 2; ++nt)
            #pragma unroll
            for (int kc = 0; kc < 2; ++kc)
                kfn[nt][kc] = load_b_frag(Kb + (size_t)(sn + nt * 16) * DKH, DKH, kc * 32, lane);
        v16bf vf[4];
        #pragma unroll
        for (int c = 0; c < 4; ++c)
            vf[c] = load_b_frag(Vb + (size_t)(c * 16) * T_, T_, sBase, lane);

        // ---- causal mask + online softmax (rows live in 16-lane half-groups)
        #pragma unroll
        for (int mt = 0; mt < 2; ++mt) {
            float corr[8];
            #pragma unroll
            for (int g = 0; g < 8; ++g) {
                int t = mBase + mt * 16 + g + half * 8;
                float v0 = (sBase + nloc      <= t) ? s[mt][0][g] : -1.0e9f;
                float v1 = (sBase + 16 + nloc <= t) ? s[mt][1][g] : -1.0e9f;
                float v = fmaxf(v0, v1);
                v = fmaxf(v, __shfl_xor(v, 1, 32));
                v = fmaxf(v, __shfl_xor(v, 2, 32));
                v = fmaxf(v, __shfl_xor(v, 4, 32));
                v = fmaxf(v, __shfl_xor(v, 8, 32));
                float nm = fmaxf(m_i[mt][g], v);
                corr[g] = __expf(m_i[mt][g] - nm);
                m_i[mt][g] = nm;
                float p0 = __expf(v0 - nm);
                float p1 = __expf(v1 - nm);
                s[mt][0][g] = p0; s[mt][1][g] = p1;
                float ps = p0 + p1;
                ps += __shfl_xor(ps, 1, 32);
                ps += __shfl_xor(ps, 2, 32);
                ps += __shfl_xor(ps, 4, 32);
                ps += __shfl_xor(ps, 8, 32);
                l_i[mt][g] = l_i[mt][g] * corr[g] + ps;
            }
            #pragma unroll
            for (int c = 0; c < 4; ++c)
                #pragma unroll
                for (int g = 0; g < 8; ++g) o[mt][c][g] *= corr[g];

            // stage P (C-layout) -> LDS row-major 32x32
            #pragma unroll
            for (int g = 0; g < 8; ++g) {
                int m = mt * 16 + g + half * 8;
                myP[m * 32 + nloc]      = (bf16_t)s[mt][0][g];
                myP[m * 32 + 16 + nloc] = (bf16_t)s[mt][1][g];
            }
        }
        asm volatile("s_wait_dscnt 0" ::: "memory");  // DS pipe in-order per wave

        // ---- O += P V  (reload P as A-fragments, V stored transposed)
        v16bf pf[2];
        #pragma unroll
        for (int mt = 0; mt < 2; ++mt)
            pf[mt] = load_a_frag(myP + mt * 16 * 32, 32, 0, lane);
        #pragma unroll
        for (int mt = 0; mt < 2; ++mt)
            #pragma unroll
            for (int c = 0; c < 4; ++c)
                o[mt][c] = wmma_bf16f32(pf[mt], vf[c], o[mt][c]);

        #pragma unroll
        for (int nt = 0; nt < 2; ++nt)
            #pragma unroll
            for (int kc = 0; kc < 2; ++kc) kf[nt][kc] = kfn[nt][kc];
    }

    // ---- normalize and write att output row-major [b*T+t, h*64+d] bf16
    #pragma unroll
    for (int mt = 0; mt < 2; ++mt) {
        #pragma unroll
        for (int g = 0; g < 8; ++g) {
            float inv = 1.0f / l_i[mt][g];
            int t = mBase + mt * 16 + g + half * 8;
            size_t row = (size_t)b * T_ + t;
            #pragma unroll
            for (int c = 0; c < 4; ++c) {
                int col = h * DKH + c * 16 + nloc;
                AttOut[row * DM + col] = (bf16_t)(o[mt][c][g] * inv);
            }
        }
    }
}

// ---------------------------------------------------------------- out proj
// Att[8192,1024]bf16 @ Wproj^T-stored[1024,1024]bf16 -> f32 d_out
__global__ void __launch_bounds__(128) proj_gemm_kernel(const bf16_t* __restrict__ A,
                                                        const bf16_t* __restrict__ Wt,
                                                        float* __restrict__ Out) {
    const int lane = threadIdx.x & 31;
    const int wave = threadIdx.x >> 5;
    const int NG = DM / 64;                         // 16 column groups
    int tile   = blockIdx.x * 4 + wave;
    int mTile  = tile / NG;
    int nGroup = tile - mTile * NG;
    int rowBase = mTile * 32;
    int colBase = nGroup * 64;

    v8f acc[2][4];
    #pragma unroll
    for (int mt = 0; mt < 2; ++mt)
        #pragma unroll
        for (int nt = 0; nt < 4; ++nt) acc[mt][nt] = zero8();

    gemm_core_32x64(A, rowBase, Wt, colBase, acc, lane);

    const int half = lane >> 4, nloc = lane & 15;
    #pragma unroll
    for (int nt = 0; nt < 4; ++nt) {
        int c = colBase + nt * 16 + nloc;
        #pragma unroll
        for (int mt = 0; mt < 2; ++mt) {
            #pragma unroll
            for (int g = 0; g < 8; ++g) {
                int r = rowBase + mt * 16 + g + half * 8;
                Out[(size_t)r * DM + c] = acc[mt][nt][g];
            }
        }
    }
}

// ---------------------------------------------------------------- launch
extern "C" void kernel_launch(void* const* d_in, const int* in_sizes, int n_in,
                              void* d_out, int out_size, void* d_ws, size_t ws_size,
                              hipStream_t stream) {
    const float* x     = (const float*)d_in[0];   // [4,2048,1024]
    const float* Wqkv  = (const float*)d_in[1];   // [1024,3072]
    const float* Wproj = (const float*)d_in[2];   // [1024,1024]
    float* out = (float*)d_out;                   // [4,2048,1024]

    // workspace layout (bf16 buffers)
    char* ws = (char*)d_ws;
    bf16_t* Xb  = (bf16_t*)ws;  ws += (size_t)BT * DM * 2;            // 16 MB
    bf16_t* Wqt = (bf16_t*)ws;  ws += (size_t)NQKV * DM * 2;          //  6 MB
    bf16_t* Wpt = (bf16_t*)ws;  ws += (size_t)DM * DM * 2;            //  2 MB
    bf16_t* Qb  = (bf16_t*)ws;  ws += (size_t)B_ * NH * T_ * DKH * 2; // 16 MB
    bf16_t* Kb  = (bf16_t*)ws;  ws += (size_t)B_ * NH * T_ * DKH * 2; // 16 MB
    bf16_t* Vtb = (bf16_t*)ws;  ws += (size_t)B_ * NH * DKH * T_ * 2; // 16 MB
    bf16_t* Att = (bf16_t*)ws;  ws += (size_t)BT * DM * 2;            // 16 MB

    // 1) precision conversion + weight transposes (tiled, coalesced)
    {
        int n4 = (BT * DM) / 4;
        cvt_f32_bf16_kernel<<<(n4 + 255) / 256, 256, 0, stream>>>((const float4*)x, (v4bf*)Xb, n4);
    }
    {
        dim3 grid(NQKV / 32, DM / 32);
        transpose_cvt_tiled<DM, NQKV><<<grid, 256, 0, stream>>>(Wqkv, Wqt);
    }
    {
        dim3 grid(DM / 32, DM / 32);
        transpose_cvt_tiled<DM, DM><<<grid, 256, 0, stream>>>(Wproj, Wpt);
    }

    // 2) QKV projection (WMMA), scatter to Q (scaled), K, V-transposed
    {
        int tiles = (BT / 32) * (NQKV / 64);   // 256 * 48 = 12288 waves
        qkv_gemm_kernel<<<tiles / 4, 128, 0, stream>>>(Xb, Wqt, Qb, Kb, Vtb);
    }

    // 3) causal flash attention (WMMA)
    {
        int waves = B_ * NH * (T_ / 32);       // 4096 waves
        attn_kernel<<<waves / 4, 128, 0, stream>>>(Qb, Kb, Vtb, Att);
    }

    // 4) output projection (WMMA) -> f32
    {
        int tiles = (BT / 32) * (DM / 64);     // 256 * 16 = 4096 waves
        proj_gemm_kernel<<<tiles / 4, 128, 0, stream>>>(Att, Wpt, out);
    }
}
